// qiao_infer_52329881534724
// MI455X (gfx1250) — compile-verified, matrix operand order checked by reading, not run on hardware
//
#include <hip/hip_runtime.h>

// ---------------- model dims ----------------
#define DMODEL 512
#define NHEAD  8
#define DKH    64
#define DFFN   2048
#define NLAYER 2
#define SEQL   128
#define BATCH  4
#define MROWS  (BATCH*SEQL)   // 512
#define LNEPS  1e-6f

typedef __attribute__((ext_vector_type(16))) _Float16 v16h;
typedef __attribute__((ext_vector_type(8)))  float    v8f;

union HFrag { v16h v; _Float16 h[16]; uint4 q[2]; };
union FAcc  { v8f  v; float    f[8]; };

__device__ __forceinline__ float wave_sum32(float v) {
#pragma unroll
  for (int m = 16; m >= 1; m >>= 1) v += __shfl_xor(v, m, 32);
  return v;
}
__device__ __forceinline__ float wave_max32(float v) {
#pragma unroll
  for (int m = 16; m >= 1; m >>= 1) v = fmaxf(v, __shfl_xor(v, m, 32));
  return v;
}

// ---------------- generic WMMA GEMM ----------------
// C[M,N] = act(alpha * A[M,K] @ B + bias + resid)
// BT==1 : B given transposed as [N,K] row-major (contiguous fragment loads)
// BT==0 : B given as [K,N] row-major (strided loads)
// One wave (32 threads) computes one 16x16 tile; batch via blockIdx.y/z strides.
struct GemmArgs {
  const _Float16* A; long sAy, sAz; int lda;
  const _Float16* B; long sBy, sBz; int ldb;
  float*          C; long sCy, sCz; int ldc;
  _Float16*     C16; long s16y, s16z; int ldc16;
  const float* bias;
  const float* resid; long sRy, sRz; int ldr;
  int M, N, K;
  float alpha;
  int relu;
};

template <int BT>
__global__ void k_gemm(GemmArgs g) {
  const int lane   = threadIdx.x;
  const int tilesN = g.N >> 4;
  const int tile   = blockIdx.x;
  const int tm = tile / tilesN, tn = tile % tilesN;

  const _Float16* A = g.A + (long)blockIdx.y * g.sAy + (long)blockIdx.z * g.sAz;
  const _Float16* B = g.B + (long)blockIdx.y * g.sBy + (long)blockIdx.z * g.sBz;

  const int row   = tm * 16 + (lane & 15);
  const int colN  = tn * 16 + (lane & 15);
  const int khalf = (lane >> 4) * 8;

  FAcc acc; acc.v = (v8f){};
  HFrag a, b;

  for (int k0 = 0; k0 < g.K; k0 += 32) {
    const int kb = k0 + khalf;
    if (row < g.M) {
      a.q[0] = *(const uint4*)(A + (long)row * g.lda + kb);
      a.q[1] = *(const uint4*)(A + (long)row * g.lda + kb + 16);
      if (k0 + 32 < g.K)  // cover first-touch latency on the A stream
        __builtin_prefetch(A + (long)row * g.lda + kb + 32, 0, 3);
    } else {
      a.q[0] = make_uint4(0u, 0u, 0u, 0u);
      a.q[1] = make_uint4(0u, 0u, 0u, 0u);
    }
    if (BT) {
      b.q[0] = *(const uint4*)(B + (long)colN * g.ldb + kb);
      b.q[1] = *(const uint4*)(B + (long)colN * g.ldb + kb + 16);
    } else {
#pragma unroll
      for (int c = 0; c < 16; ++c) {
        const int k = kb + c + ((c & 8) ? 8 : 0);
        b.h[c] = B[(long)k * g.ldb + colN];
      }
    }
    acc.v = __builtin_amdgcn_wmma_f32_16x16x32_f16(false, a.v, false, b.v,
                                                   (short)0, acc.v, false, false);
  }

  float*     Cp  = g.C   ? g.C   + (long)blockIdx.y * g.sCy + (long)blockIdx.z * g.sCz : nullptr;
  _Float16*  C16 = g.C16 ? g.C16 + (long)blockIdx.y * g.s16y + (long)blockIdx.z * g.s16z : nullptr;
  const float* R = g.resid ? g.resid + (long)blockIdx.y * g.sRy + (long)blockIdx.z * g.sRz : nullptr;
  const float bn = g.bias ? g.bias[colN] : 0.f;
  const int mBase = tm * 16 + ((lane >> 4) << 3);

  for (int r = 0; r < 8; ++r) {
    const int m = mBase + r;
    if (m >= g.M) break;
    float val = acc.f[r] * g.alpha + bn;
    if (R) val += R[(long)m * g.ldr + colN];
    if (g.relu) val = val > 0.f ? val : 0.f;
    if (Cp)  Cp[(long)m * g.ldc + colN] = val;
    if (C16) C16[(long)m * g.ldc16 + colN] = (_Float16)val;
  }
}

// ---------------- fused ptd kernel (dominant ~70 GFLOP) ----------------
// logits[b,o,i] = sum_n tanh( sum_k A[b,i,k] * S[b,o,k] * W[k,n] ) * vvec[n]
// W supplied transposed f16 [Dn,PTD_DK]; S folded into the B fragment at load.
// Block = 256 (8 waves) per (b, o, 16-row i-tile). A-tile (16x512, 16KB) and
// the scale vector (1KB) are staged in LDS once and reused by all 32 n-tiles
// (32x global-traffic reduction vs. per-wave streaming); B streams from the
// L2-resident weight. Cross-wave n-reduction via LDS float atomics.
#define PTD_DK 512
__global__ void k_ptd(const _Float16* A, long sAb,
                      const _Float16* WT,
                      const _Float16* S, long sSb,
                      const float* vvec, float* out, int Dn) {
  __shared__ uint4 ldsAq[16 * PTD_DK / 8];   // 16KB
  __shared__ uint4 ldsSq[PTD_DK / 8];        // 1KB
  __shared__ float rowsum[16];
  _Float16* ldsA = (_Float16*)ldsAq;
  _Float16* ldsS = (_Float16*)ldsSq;

  const int tid  = threadIdx.x;
  const int lane = tid & 31, wv = tid >> 5;
  const int b = blockIdx.z, o = blockIdx.y, tm = blockIdx.x;
  const int Linner = gridDim.x * 16;

  const _Float16* Ab = A + (long)b * sAb + (long)tm * 16 * PTD_DK;
  const _Float16* Sb = S + (long)b * sSb + (long)o * PTD_DK;

  // stage A tile (1024 x 16B) and scale vector (64 x 16B)
  {
    const uint4* gA = (const uint4*)Ab;
    for (int c = tid; c < 16 * PTD_DK / 8; c += 256) ldsAq[c] = gA[c];
    const uint4* gS = (const uint4*)Sb;
    if (tid < PTD_DK / 8) ldsSq[tid] = gS[tid];
    if (tid < 16) rowsum[tid] = 0.f;
  }
  __syncthreads();

  const int rowl  = lane & 15;
  const int khalf = (lane >> 4) * 8;

  float partial[8];
#pragma unroll
  for (int r = 0; r < 8; ++r) partial[r] = 0.f;

  HFrag a, bf, sc;
  for (int tn = wv; tn < (Dn >> 4); tn += 8) {
    const int n = tn * 16 + rowl;
    FAcc acc; acc.v = (v8f){};
    for (int k0 = 0; k0 < PTD_DK; k0 += 32) {
      const int kb = k0 + khalf;
      a.q[0]  = *(const uint4*)(ldsA + rowl * PTD_DK + kb);
      a.q[1]  = *(const uint4*)(ldsA + rowl * PTD_DK + kb + 16);
      bf.q[0] = *(const uint4*)(WT + (long)n * PTD_DK + kb);
      bf.q[1] = *(const uint4*)(WT + (long)n * PTD_DK + kb + 16);
      sc.q[0] = *(const uint4*)(ldsS + kb);
      sc.q[1] = *(const uint4*)(ldsS + kb + 16);
#pragma unroll
      for (int c = 0; c < 16; ++c) bf.h[c] = bf.h[c] * sc.h[c];
      acc.v = __builtin_amdgcn_wmma_f32_16x16x32_f16(false, a.v, false, bf.v,
                                                     (short)0, acc.v, false, false);
    }
    const float vn = vvec[n];
#pragma unroll
    for (int r = 0; r < 8; ++r) partial[r] += tanhf(acc.f[r]) * vn;
  }
  // reduce over the 16 lanes of each half (N within tile)
#pragma unroll
  for (int r = 0; r < 8; ++r) {
    float v = partial[r];
#pragma unroll
    for (int m = 8; m >= 1; m >>= 1) v += __shfl_xor(v, m, 32);
    partial[r] = v;
  }
  if ((lane & 15) == 0) {
    const int mhalf = (lane >> 4) << 3;
#pragma unroll
    for (int r = 0; r < 8; ++r) atomicAdd(&rowsum[mhalf + r], partial[r]);
  }
  __syncthreads();
  if (tid < 16)
    out[((long)b * gridDim.y + o) * Linner + tm * 16 + tid] = rowsum[tid];
}

// ---------------- pairwise tanh reduce (factorized perspectives) ----------------
// out[b,o,i] = sum_d tanhf(X[b,i,d] + sign*Y[b,o,d]) * vec[d]
// grid = (Linner/8, Louter, B), block = 256 (8 waves, one (o,i) per wave)
__global__ void k_tanh_reduce(const float* X, const float* Y, const float* vec,
                              float* out, int Dm, float sign) {
  const int wv = threadIdx.x >> 5, lane = threadIdx.x & 31;
  const int wpb = blockDim.x >> 5;
  const int i = blockIdx.x * wpb + wv;
  const int o = blockIdx.y, b = blockIdx.z;
  const int Linner = gridDim.x * wpb;
  const float* Xp = X + ((long)b * Linner + i) * Dm;
  const float* Yp = Y + ((long)b * gridDim.y + o) * Dm;
  float s = 0.f;
  for (int d = lane; d < Dm; d += 32) s += tanhf(Xp[d] + sign * Yp[d]) * vec[d];
  s = wave_sum32(s);
  if (lane == 0) out[((long)b * gridDim.y + o) * Linner + i] = s;
}

// ---------------- misc kernels ----------------
__global__ void k_embed(const int* tok, const float* emb, float* x, int Lk, int Dm) {
  const long idx = (long)blockIdx.x * blockDim.x + threadIdx.x;
  const int d = (int)(idx % Dm);
  const long bl = idx / Dm;
  const int l = (int)(bl % Lk);
  const int t = tok[bl];
  const int d2 = d & ~1;
  const float div = __expf(-logf(10000.f) * (float)d2 / (float)Dm);
  const float ang = (float)l * div;
  const float pe = (d & 1) ? cosf(ang) : sinf(ang);
  x[idx] = emb[(long)t * Dm + d] * sqrtf((float)Dm) + pe;
}

__global__ void k_w_to_f16t(const float* W, _Float16* WT, int Kd, int Nd) {
  const long idx = (long)blockIdx.x * blockDim.x + threadIdx.x;
  if (idx >= (long)Kd * Nd) return;
  const int n = (int)(idx % Nd);
  const long k = idx / Nd;
  WT[(long)n * Kd + k] = (_Float16)W[idx];
}

__global__ void k_layernorm(const float* x, const float* gamma, const float* beta,
                            float* y, _Float16* y16, int Dm) {
  const long row = blockIdx.x;
  const float* p = x + row * (long)Dm;
  __shared__ float red[4];
  const int lane = threadIdx.x & 31, wv = threadIdx.x >> 5;
  float s = 0.f;
  for (int j = threadIdx.x; j < Dm; j += blockDim.x) s += p[j];
  s = wave_sum32(s);
  if (lane == 0) red[wv] = s;
  __syncthreads();
  const float mean = (red[0] + red[1] + red[2] + red[3]) / (float)Dm;
  __syncthreads();
  float v = 0.f;
  for (int j = threadIdx.x; j < Dm; j += blockDim.x) { float d = p[j] - mean; v += d * d; }
  v = wave_sum32(v);
  if (lane == 0) red[wv] = v;
  __syncthreads();
  const float var = (red[0] + red[1] + red[2] + red[3]) / (float)Dm;
  const float inv = rsqrtf(var + LNEPS);
  for (int j = threadIdx.x; j < Dm; j += blockDim.x) {
    const float o = gamma[j] * (p[j] - mean) * inv + beta[j];
    y[row * (long)Dm + j] = o;
    y16[row * (long)Dm + j] = (_Float16)o;
  }
}

// row softmax in-place + optional f16 mirror; block = 128 (4 waves), one row per block
__global__ void k_softmax_row(float* x, _Float16* x16, int n) {
  const long row = blockIdx.x;
  float* p = x + row * (long)n;
  __shared__ float red[4];
  const int lane = threadIdx.x & 31, wv = threadIdx.x >> 5;
  float mx = -1e30f;
  for (int j = threadIdx.x; j < n; j += blockDim.x) mx = fmaxf(mx, p[j]);
  mx = wave_max32(mx);
  if (lane == 0) red[wv] = mx;
  __syncthreads();
  mx = fmaxf(fmaxf(red[0], red[1]), fmaxf(red[2], red[3]));
  __syncthreads();
  float sum = 0.f;
  for (int j = threadIdx.x; j < n; j += blockDim.x) { float e = __expf(p[j] - mx); p[j] = e; sum += e; }
  sum = wave_sum32(sum);
  if (lane == 0) red[wv] = sum;
  __syncthreads();
  const float inv = 1.f / (red[0] + red[1] + red[2] + red[3]);
  for (int j = threadIdx.x; j < n; j += blockDim.x) {
    const float v = p[j] * inv;
    p[j] = v;
    if (x16) x16[row * (long)n + j] = (_Float16)v;
  }
}

// scores[b,l] = t[b,l,:] . v   (grid (L,B), block 32)
__global__ void k_pool_score(const float* t, const float* v, float* sj, int Lk, int Dm) {
  const int l = blockIdx.x, b = blockIdx.y, lane = threadIdx.x;
  const float* p = t + ((long)b * Lk + l) * Dm;
  float s = 0.f;
  for (int d = lane; d < Dm; d += 32) s += p[d] * v[d];
  s = wave_sum32(s);
  if (lane == 0) sj[(long)b * Lk + l] = s;
}

// rc[b,k] = sum_l sj[b,l] * agg[b,l,k]
__global__ void k_weighted_sum(const float* sj, const float* agg, float* rc, int Lk, int Kd) {
  const long idx = (long)blockIdx.x * blockDim.x + threadIdx.x;
  const int k = (int)(idx % Kd);
  const int b = (int)(idx / Kd);
  float s = 0.f;
  for (int l = 0; l < Lk; ++l) s += sj[(long)b * Lk + l] * agg[((long)b * Lk + l) * Kd + k];
  rc[idx] = s;
}

// pair = [rc, rc_, rc*rc_, |rc-rc_|] in f16
__global__ void k_pair(const float* rc, const float* rc_, _Float16* pair16, int Kd) {
  const long idx = (long)blockIdx.x * blockDim.x + threadIdx.x;
  const int b = (int)(idx / (4 * (long)Kd));
  const int r = (int)(idx % (4 * (long)Kd));
  const int q = r / Kd, k = r % Kd;
  const float a = rc[(long)b * Kd + k], c = rc_[(long)b * Kd + k];
  float v;
  if (q == 0) v = a;
  else if (q == 1) v = c;
  else if (q == 2) v = a * c;
  else v = fabsf(a - c);
  pair16[idx] = (_Float16)v;
}

__global__ void k_final(const float* hid, const float* w2, const float* b2, float* out, int Dm) {
  const int b = blockIdx.x, lane = threadIdx.x;
  float s = 0.f;
  for (int d = lane; d < Dm; d += 32) s += hid[(long)b * Dm + d] * w2[d];
  s = wave_sum32(s);
  if (lane == 0) {
    const float z = s + b2[0];
    out[b] = 1.f / (1.f + __expf(-z));
  }
}

// ---------------- host side ----------------
static void gemm(hipStream_t st, bool bt,
                 const _Float16* A, long sAy, long sAz, int lda,
                 const _Float16* B, long sBy, long sBz, int ldb,
                 float* C, long sCy, long sCz, int ldc,
                 _Float16* C16, long s16y, long s16z, int ldc16,
                 const float* bias, const float* resid, long sRy, long sRz, int ldr,
                 int M, int N, int K, float alpha, int relu, int gy, int gz) {
  GemmArgs g{A, sAy, sAz, lda, B, sBy, sBz, ldb, C, sCy, sCz, ldc,
             C16, s16y, s16z, ldc16, bias, resid, sRy, sRz, ldr, M, N, K, alpha, relu};
  dim3 grid((unsigned)(((M + 15) / 16) * (N / 16)), (unsigned)gy, (unsigned)gz);
  dim3 block(32, 1, 1);
  if (bt) k_gemm<1><<<grid, block, 0, st>>>(g);
  else    k_gemm<0><<<grid, block, 0, st>>>(g);
}

// single-batch convenience: B transposed [N,K], ldc = N
static void gemm1(hipStream_t st, const _Float16* A, int lda, const _Float16* WT, int ldwt,
                  float* C, _Float16* C16, const float* bias, const float* resid,
                  int M, int N, int K, int relu) {
  gemm(st, true, A, 0, 0, lda, WT, 0, 0, ldwt, C, 0, 0, N, C16, 0, 0, N,
       bias, resid, 0, 0, N, M, N, K, 1.f, relu, 1, 1);
}

// flattened-input indices (JAX tree order: dict keys ASCII-sorted)
enum {
  IN_COMM = 0, IN_Vm = 1, IN_Vm_ = 2, IN_Wb = 3, IN_Wb_ = 4, IN_Wc1 = 5, IN_Wc1_ = 6,
  IN_Wc1_p = 7, IN_Wc1_p_ = 8, IN_Wc2 = 9, IN_Wc2_ = 10, IN_Wd = 11, IN_Wd_ = 12,
  IN_Wm = 13, IN_Wm_ = 14, IN_EMB = 15, IN_LAYER0 = 16,
  IN_LNF_B = 48, IN_LNF_G = 49, IN_MLP_W1 = 50, IN_MLP_W2 = 51, IN_MLP_B1 = 52, IN_MLP_B2 = 53,
  IN_VC = 54, IN_VC_ = 55, IN_VC_P = 56, IN_VC_P_ = 57, IN_VD = 58, IN_VD_ = 59, IN_POST = 60
};
// per-layer offsets (ASCII sorted keys)
enum { LW1 = 0, LW2 = 1, LWK = 2, LWO = 3, LWQ = 4, LWV = 5, LB1 = 6, LB2 = 7, LBK = 8,
       LBO = 9, LBQ = 10, LBV = 11, LLN1B = 12, LLN1G = 13, LLN2B = 14, LLN2G = 15 };

extern "C" void kernel_launch(void* const* d_in, const int* in_sizes, int n_in,
                              void* d_out, int out_size, void* d_ws, size_t ws_size,
                              hipStream_t stream) {
  (void)in_sizes; (void)n_in; (void)out_size; (void)ws_size;
  const hipStream_t st = stream;
  auto F = [&](int i) { return (const float*)d_in[i]; };
  auto FL = [&](int l, int o) { return (const float*)d_in[IN_LAYER0 + 16 * l + o]; };

  // ---- workspace bump allocator ----
  char* ws = (char*)d_ws;
  size_t off = 0;
  auto alloc = [&](size_t bytes) -> void* {
    size_t o = (off + 255) & ~(size_t)255;
    off = o + bytes;
    return (void*)(ws + o);
  };
  auto allocF = [&](size_t n) { return (float*)alloc(n * sizeof(float)); };
  auto allocH = [&](size_t n) { return (_Float16*)alloc(n * sizeof(_Float16)); };

  // ---- f16-transposed weights ----
  auto prepT = [&](const float* W, int K, int N) {
    _Float16* WT = allocH((size_t)K * N);
    long e = (long)K * N;
    k_w_to_f16t<<<(unsigned)((e + 255) / 256), 256, 0, st>>>(W, WT, K, N);
    return WT;
  };
  _Float16 *WqT[NLAYER], *WkT[NLAYER], *WvT[NLAYER], *WoT[NLAYER], *W1T[NLAYER], *W2T[NLAYER];
  for (int l = 0; l < NLAYER; ++l) {
    WqT[l] = prepT(FL(l, LWQ), DMODEL, DMODEL);
    WkT[l] = prepT(FL(l, LWK), DMODEL, DMODEL);
    WvT[l] = prepT(FL(l, LWV), DMODEL, DMODEL);
    WoT[l] = prepT(FL(l, LWO), DMODEL, DMODEL);
    W1T[l] = prepT(FL(l, LW1), DMODEL, DFFN);
    W2T[l] = prepT(FL(l, LW2), DFFN, DMODEL);
  }
  _Float16* Wc1T  = prepT(F(IN_Wc1),  DMODEL, DMODEL);
  _Float16* Wc2T  = prepT(F(IN_Wc2),  DMODEL, DMODEL);
  _Float16* WbT   = prepT(F(IN_Wb),   DMODEL, DMODEL);
  _Float16* WdT   = prepT(F(IN_Wd),   DMODEL, DMODEL);
  _Float16* WmT   = prepT(F(IN_Wm),   DMODEL, DMODEL);
  _Float16* Wc1_T = prepT(F(IN_Wc1_), DMODEL, DMODEL);
  _Float16* Wc2_T = prepT(F(IN_Wc2_), DMODEL, DMODEL);
  _Float16* Wb_T  = prepT(F(IN_Wb_),  DMODEL, DMODEL);
  _Float16* Wd_T  = prepT(F(IN_Wd_),  DMODEL, DMODEL);
  _Float16* Wm_T  = prepT(F(IN_Wm_),  DMODEL, DMODEL);
  _Float16* Wc1pT  = prepT(F(IN_Wc1_p),  4 * DMODEL, DMODEL);
  _Float16* Wc1p_T = prepT(F(IN_Wc1_p_), 4 * DMODEL, DMODEL);
  _Float16* mlpW1T = prepT(F(IN_MLP_W1), 16 * DMODEL, DMODEL);

  // ---- activation scratch ----
  const size_t MD = (size_t)MROWS * DMODEL;
  float*     x      = allocF(MD);
  float*     hbuf   = allocF(MD);
  _Float16*  h16    = allocH(MD);
  float*     qf     = allocF(MD);  _Float16* q16 = allocH(MD);
  float*     kf     = allocF(MD);  _Float16* k16 = allocH(MD);
  float*     vf     = allocF(MD);  _Float16* v16 = allocH(MD);
  float*     scores = allocF((size_t)BATCH * NHEAD * SEQL * SEQL);
  _Float16*  a16    = allocH((size_t)BATCH * NHEAD * SEQL * SEQL);
  float*     attno  = allocF(MD);  _Float16* attno16 = allocH(MD);
  float*     ffn    = allocF((size_t)MROWS * DFFN);
  _Float16*  ffn16  = allocH((size_t)MROWS * DFFN);
  float*     hp  = allocF(MD);  _Float16* hp16 = allocH(MD);
  float*     hc  = allocF(MD);  _Float16* hc16 = allocH(MD);

  float* Xc1  = allocF(MD);
  float* Xc2  = allocF(MD);
  float* Ybf  = allocF(MD);  _Float16* Yb16  = allocH(MD);
  float* Pm   = allocF(MD);
  float* Qm   = allocF(MD);
  float* Xc1_ = allocF(MD);
  float* Xc2_ = allocF(MD);  _Float16* Xc2_16 = allocH(MD);
  float* Yb_f = allocF(MD);  _Float16* Yb_16  = allocH(MD);
  float* Pm_  = allocF(MD);
  float* Qm_  = allocF(MD);

  float*    logits = allocF((size_t)BATCH * SEQL * SEQL);
  _Float16* lg16   = allocH((size_t)BATCH * SEQL * SEQL);
  float*    agg_p  = allocF((size_t)MROWS * 4 * DMODEL);
  _Float16* aggp16 = allocH((size_t)MROWS * 4 * DMODEL);
  float*    agg_c  = allocF((size_t)MROWS * 4 * DMODEL);
  _Float16* aggc16 = allocH((size_t)MROWS * 4 * DMODEL);
  float*    tpool  = allocF(MD);
  float*    sj     = allocF((size_t)BATCH * SEQL);
  float*    sj_    = allocF((size_t)BATCH * SEQL);
  float*    rc     = allocF((size_t)BATCH * 4 * DMODEL);
  float*    rc_    = allocF((size_t)BATCH * 4 * DMODEL);
  _Float16* pair16 = allocH((size_t)BATCH * 16 * DMODEL);
  float*    hid    = allocF((size_t)BATCH * DMODEL);

  const long sHB = (long)SEQL * DMODEL;          // batch stride of [B,L,D]
  const long sSC = (long)SEQL * SEQL;            // per-head stride of scores
  const long sSB = (long)NHEAD * SEQL * SEQL;    // per-batch stride of scores

  // ---- encoder (shared weights), run twice ----
  auto encoder = [&](const int* tok, float* hfin, _Float16* hfin16) {
    k_embed<<<(unsigned)(MD / 256), 256, 0, st>>>(tok, F(IN_EMB), x, SEQL, DMODEL);
    for (int l = 0; l < NLAYER; ++l) {
      k_layernorm<<<MROWS, 128, 0, st>>>(x, FL(l, LLN1G), FL(l, LLN1B), hbuf, h16, DMODEL);
      gemm1(st, h16, DMODEL, WqT[l], DMODEL, qf, q16, FL(l, LBQ), nullptr, MROWS, DMODEL, DMODEL, 0);
      gemm1(st, h16, DMODEL, WkT[l], DMODEL, kf, k16, FL(l, LBK), nullptr, MROWS, DMODEL, DMODEL, 0);
      gemm1(st, h16, DMODEL, WvT[l], DMODEL, vf, v16, FL(l, LBV), nullptr, MROWS, DMODEL, DMODEL, 0);
      // scores[b,h] = q @ k^T / sqrt(dk)
      gemm(st, true, q16, DKH, sHB, DMODEL, k16, DKH, sHB, DMODEL,
           scores, sSC, sSB, SEQL, nullptr, 0, 0, 0,
           nullptr, nullptr, 0, 0, 0, SEQL, SEQL, DKH, 0.125f, 0, NHEAD, BATCH);
      k_softmax_row<<<BATCH * NHEAD * SEQL, 128, 0, st>>>(scores, a16, SEQL);
      // o = a @ v
      gemm(st, false, a16, sSC, sSB, SEQL, v16, DKH, sHB, DMODEL,
           attno, DKH, sHB, DMODEL, attno16, DKH, sHB, DMODEL,
           nullptr, nullptr, 0, 0, 0, SEQL, DKH, SEQL, 1.f, 0, NHEAD, BATCH);
      // x += o @ Wo + bo
      gemm1(st, attno16, DMODEL, WoT[l], DMODEL, x, nullptr, FL(l, LBO), x, MROWS, DMODEL, DMODEL, 0);
      k_layernorm<<<MROWS, 128, 0, st>>>(x, FL(l, LLN2G), FL(l, LLN2B), hbuf, h16, DMODEL);
      gemm1(st, h16, DMODEL, W1T[l], DMODEL, ffn, ffn16, FL(l, LB1), nullptr, MROWS, DFFN, DMODEL, 1);
      gemm1(st, ffn16, DFFN, W2T[l], DFFN, x, nullptr, FL(l, LB2), x, MROWS, DMODEL, DFFN, 0);
    }
    k_layernorm<<<MROWS, 128, 0, st>>>(x, F(IN_LNF_G), F(IN_LNF_B), hfin, hfin16, DMODEL);
  };
  encoder((const int*)d_in[IN_POST], hp, hp16);
  encoder((const int*)d_in[IN_COMM], hc, hc16);

  // ---- perspective projections ----
  gemm1(st, hp16, DMODEL, Wc1T,  DMODEL, Xc1,  nullptr, nullptr, nullptr, MROWS, DMODEL, DMODEL, 0);
  gemm1(st, hc16, DMODEL, Wc2T,  DMODEL, Xc2,  nullptr, nullptr, nullptr, MROWS, DMODEL, DMODEL, 0);
  gemm1(st, hp16, DMODEL, WbT,   DMODEL, Ybf,  Yb16,    nullptr, nullptr, MROWS, DMODEL, DMODEL, 0);
  gemm1(st, hp16, DMODEL, WmT,   DMODEL, Pm,   nullptr, nullptr, nullptr, MROWS, DMODEL, DMODEL, 0);
  gemm1(st, hc16, DMODEL, WmT,   DMODEL, Qm,   nullptr, nullptr, nullptr, MROWS, DMODEL, DMODEL, 0);
  gemm1(st, hc16, DMODEL, Wc1_T, DMODEL, Xc1_, nullptr, nullptr, nullptr, MROWS, DMODEL, DMODEL, 0);
  gemm1(st, hp16, DMODEL, Wc2_T, DMODEL, Xc2_, Xc2_16,  nullptr, nullptr, MROWS, DMODEL, DMODEL, 0);
  gemm1(st, hc16, DMODEL, Wb_T,  DMODEL, Yb_f, Yb_16,   nullptr, nullptr, MROWS, DMODEL, DMODEL, 0);
  gemm1(st, hc16, DMODEL, Wm_T,  DMODEL, Pm_,  nullptr, nullptr, nullptr, MROWS, DMODEL, DMODEL, 0);
  gemm1(st, hp16, DMODEL, Wm_T,  DMODEL, Qm_,  nullptr, nullptr, nullptr, MROWS, DMODEL, DMODEL, 0);

  // softmax(last axis) then apply attention over ctx into agg slice
  auto apply = [&](const _Float16* ctx16, float* aggBase, _Float16* agg16Base, int colOff) {
    k_softmax_row<<<BATCH * SEQL, 128, 0, st>>>(logits, lg16, SEQL);
    gemm(st, false, lg16, 0, (long)SEQL * SEQL, SEQL,
         ctx16, 0, sHB, DMODEL,
         aggBase + colOff, 0, (long)SEQL * 4 * DMODEL, 4 * DMODEL,
         agg16Base + colOff, 0, (long)SEQL * 4 * DMODEL, 4 * DMODEL,
         nullptr, nullptr, 0, 0, 0, SEQL, DMODEL, SEQL, 1.f, 0, 1, BATCH);
  };
  const dim3 gTR(16, SEQL, BATCH), bTR(256);
  const dim3 gPD(SEQL / 16, SEQL, BATCH), bPD(256);

  // ---- p -> c perspectives (logits[b,j,i], ctx = hp) ----
  k_tanh_reduce<<<gTR, bTR, 0, st>>>(Xc1, Xc2, F(IN_VC), logits, DMODEL, 1.f);
  apply(hp16, agg_p, aggp16, 0);
  gemm(st, true, hc16, 0, sHB, DMODEL, Yb16, 0, sHB, DMODEL,
       logits, 0, (long)SEQL * SEQL, SEQL, nullptr, 0, 0, 0,
       nullptr, nullptr, 0, 0, 0, SEQL, SEQL, DMODEL, 1.f, 0, 1, BATCH);
  apply(hp16, agg_p, aggp16, DMODEL);
  k_ptd<<<gPD, bPD, 0, st>>>(hp16, sHB, WdT, hc16, sHB, F(IN_VD), logits, DMODEL);
  apply(hp16, agg_p, aggp16, 2 * DMODEL);
  k_tanh_reduce<<<gTR, bTR, 0, st>>>(Pm, Qm, F(IN_Vm), logits, DMODEL, -1.f);
  apply(hp16, agg_p, aggp16, 3 * DMODEL);

  // ---- c -> p perspectives (logits[b,i,j], ctx = hc) ----
  k_tanh_reduce<<<gTR, bTR, 0, st>>>(Xc1_, Xc2_, F(IN_VC_), logits, DMODEL, 1.f);
  apply(hc16, agg_c, aggc16, 0);
  gemm(st, true, hp16, 0, sHB, DMODEL, Yb_16, 0, sHB, DMODEL,
       logits, 0, (long)SEQL * SEQL, SEQL, nullptr, 0, 0, 0,
       nullptr, nullptr, 0, 0, 0, SEQL, SEQL, DMODEL, 1.f, 0, 1, BATCH);
  apply(hc16, agg_c, aggc16, DMODEL);
  k_ptd<<<gPD, bPD, 0, st>>>(hc16, sHB, Wd_T, Xc2_16, sHB, F(IN_VD_), logits, DMODEL);
  apply(hc16, agg_c, aggc16, 2 * DMODEL);
  k_tanh_reduce<<<gTR, bTR, 0, st>>>(Pm_, Qm_, F(IN_Vm_), logits, DMODEL, -1.f);
  apply(hc16, agg_c, aggc16, 3 * DMODEL);

  // ---- pooling ----
  gemm1(st, aggp16, 4 * DMODEL, Wc1pT, 4 * DMODEL, tpool, nullptr, nullptr, nullptr,
        MROWS, DMODEL, 4 * DMODEL, 0);
  k_pool_score<<<dim3(SEQL, BATCH), 32, 0, st>>>(tpool, F(IN_VC_P), sj, SEQL, DMODEL);
  k_softmax_row<<<BATCH, 128, 0, st>>>(sj, nullptr, SEQL);
  k_weighted_sum<<<(BATCH * 4 * DMODEL) / 256, 256, 0, st>>>(sj, agg_p, rc, SEQL, 4 * DMODEL);

  gemm1(st, aggc16, 4 * DMODEL, Wc1p_T, 4 * DMODEL, tpool, nullptr, nullptr, nullptr,
        MROWS, DMODEL, 4 * DMODEL, 0);
  k_pool_score<<<dim3(SEQL, BATCH), 32, 0, st>>>(tpool, F(IN_VC_P_), sj_, SEQL, DMODEL);
  k_softmax_row<<<BATCH, 128, 0, st>>>(sj_, nullptr, SEQL);
  k_weighted_sum<<<(BATCH * 4 * DMODEL) / 256, 256, 0, st>>>(sj_, agg_c, rc_, SEQL, 4 * DMODEL);

  // ---- final MLP ----
  k_pair<<<(BATCH * 16 * DMODEL) / 256, 256, 0, st>>>(rc, rc_, pair16, 4 * DMODEL);
  gemm1(st, pair16, 16 * DMODEL, mlpW1T, 16 * DMODEL, hid, nullptr, F(IN_MLP_B1), nullptr,
        BATCH, DMODEL, 16 * DMODEL, 1);
  k_final<<<BATCH, 32, 0, st>>>(hid, F(IN_MLP_W2), F(IN_MLP_B2), (float*)d_out, DMODEL);
}
